// Sp_GalerkinAttention_7739531068085
// MI455X (gfx1250) — matrix operand order repaired and checked by reading, not compiled, not run on hardware
//
#include <hip/hip_runtime.h>

typedef __attribute__((ext_vector_type(16))) _Float16 v16h;
typedef __attribute__((ext_vector_type(8)))  _Float16 v8h;
typedef __attribute__((ext_vector_type(8)))  float    v8f;

#define NBATCH   4
#define SLEN     8192
#define DHID     512
#define NHEAD    16
#define DHEAD    32
#define CCAT     34      // pos(2) + head dim(32)
#define CPAD     48      // padded per-head concat dim for WMMA tiles
#define KTVLD    64      // padded ktv leading dim (K-dim for build_m)
#define CHFULL   544     // 16 * 34
#define NORM_EPS 1e-5f

static __device__ inline v8f wmma16(v16h a, v16h b, v8f c) {
  return __builtin_amdgcn_wmma_f32_16x16x32_f16(
      false, a, false, b, (short)0, c, false, false);
}

// Async global->LDS 16-byte copy (CDNA5, tracked by ASYNCcnt).
// LDS aperture check uses addr[63:32] only, so the low 32 bits of a flat
// shared-pointer are the LDS byte offset.
static __device__ __forceinline__ void async_b128(unsigned ldsoff,
                                                  const void* gptr) {
  unsigned long long ga = (unsigned long long)(uintptr_t)gptr;
  asm volatile("global_load_async_to_lds_b128 %0, %1, off"
               :
               : "v"(ldsoff), "v"(ga)
               : "memory");
}
static __device__ __forceinline__ void wait_async0() {
  asm volatile("s_wait_asynccnt 0x0" ::: "memory");
}

// ---------------------------------------------------------------------------
// init kernels
// ---------------------------------------------------------------------------
__global__ void zero_ktv_kernel(float* __restrict__ ktv) {
  int i = blockIdx.x * 256 + threadIdx.x;
  if (i < NBATCH * NHEAD * CPAD * KTVLD) ktv[i] = 0.0f;
}

__global__ void init_pos_kernel(const float* __restrict__ pos,
                                _Float16* __restrict__ qcat,
                                _Float16* __restrict__ kcat,
                                _Float16* __restrict__ vcat) {
  int idx = blockIdx.x * 256 + threadIdx.x;          // 0..N*S-1
  if (idx >= NBATCH * SLEN) return;
  int n = idx >> 13, s = idx & (SLEN - 1);
  _Float16 h0 = (_Float16)pos[(size_t)idx * 2 + 0];
  _Float16 h1 = (_Float16)pos[(size_t)idx * 2 + 1];
  _Float16* q = qcat + (size_t)idx * CHFULL;
#pragma unroll
  for (int h = 0; h < NHEAD; ++h) { q[h * CCAT] = h0; q[h * CCAT + 1] = h1; }
#pragma unroll
  for (int h = 0; h < NHEAD; ++h) {
    size_t b = (((size_t)n * NHEAD + h) * SLEN + s) * CPAD;
    kcat[b] = h0; kcat[b + 1] = h1;
    vcat[b] = h0; vcat[b + 1] = h1;
#pragma unroll
    for (int c = CCAT; c < CPAD; ++c) {
      kcat[b + c] = (_Float16)0.0f;
      vcat[b + c] = (_Float16)0.0f;
    }
  }
}

// ---------------------------------------------------------------------------
// Projection GEMM: Y[128,64] tile = X[128,512] @ W[512,64] (+bias, opt LN)
// block = 256 threads (8 waves); wave w -> rows 16w..16w+15, all 64 cols.
// fp32 inputs must be converted to f16, so staging goes through VGPRs here.
// ---------------------------------------------------------------------------
template <int DO_LN>
__global__ __launch_bounds__(256) void proj_kernel(
    const float* __restrict__ X, const float* __restrict__ W,
    const float* __restrict__ bias, const float* __restrict__ lnw,
    const float* __restrict__ lnb, _Float16* __restrict__ out) {
  __shared__ __align__(16) _Float16 ldsA[128][32];
  __shared__ __align__(16) _Float16 ldsBt[64][32];   // [col][k]

  const int tid  = threadIdx.x;
  const int lane = tid & 31;
  const int w    = tid >> 5;
  const int nlo  = lane & 15;
  const int half = lane >> 4;
  const int n    = blockIdx.z;
  const int s0   = blockIdx.x * 128;
  const int c0   = blockIdx.y * 64;

  v8f acc[4] = {};

  for (int k0 = 0; k0 < DHID; k0 += 32) {
    // stage A tile (fp32 -> f16): row-major [128][32]
    {
      int row = tid >> 1;
      int ks  = (tid & 1) * 16;
      const float4* src =
          (const float4*)&X[((size_t)n * SLEN + s0 + row) * DHID + k0 + ks];
      float4 f0 = src[0], f1 = src[1], f2 = src[2], f3 = src[3];
      _Float16* d = &ldsA[row][ks];
      d[0]=(_Float16)f0.x; d[1]=(_Float16)f0.y; d[2]=(_Float16)f0.z; d[3]=(_Float16)f0.w;
      d[4]=(_Float16)f1.x; d[5]=(_Float16)f1.y; d[6]=(_Float16)f1.z; d[7]=(_Float16)f1.w;
      d[8]=(_Float16)f2.x; d[9]=(_Float16)f2.y; d[10]=(_Float16)f2.z; d[11]=(_Float16)f2.w;
      d[12]=(_Float16)f3.x; d[13]=(_Float16)f3.y; d[14]=(_Float16)f3.z; d[15]=(_Float16)f3.w;
    }
    // stage B tile transposed: ldsBt[col][k]
    {
      int kk = tid >> 4;             // 0..15
      int cs = (tid & 15) * 4;       // 0..60
#pragma unroll
      for (int it = 0; it < 2; ++it) {
        int k = kk + it * 16;
        float4 f = *(const float4*)&W[(size_t)(k0 + k) * DHID + c0 + cs];
        ldsBt[cs + 0][k] = (_Float16)f.x;
        ldsBt[cs + 1][k] = (_Float16)f.y;
        ldsBt[cs + 2][k] = (_Float16)f.z;
        ldsBt[cs + 3][k] = (_Float16)f.w;
      }
    }
    __syncthreads();

    const int hoff8 = half * 8;
    v8h alo = *(const v8h*)&ldsA[16 * w + nlo][hoff8];
    v8h ahi = *(const v8h*)&ldsA[16 * w + nlo][16 + hoff8];
    v16h a;
#pragma unroll
    for (int i = 0; i < 8; ++i) { a[i] = alo[i]; a[8 + i] = ahi[i]; }
#pragma unroll
    for (int t = 0; t < 4; ++t) {
      v8h blo = *(const v8h*)&ldsBt[16 * t + nlo][half * 8];
      v8h bhi = *(const v8h*)&ldsBt[16 * t + nlo][16 + half * 8];
      v16h bb;
#pragma unroll
      for (int i = 0; i < 8; ++i) { bb[i] = blo[i]; bb[8 + i] = bhi[i]; }
      acc[t] = wmma16(a, bb, acc[t]);
    }
    __syncthreads();
  }

  // epilogue
  float bv[4], lwv[4], lbv[4];
#pragma unroll
  for (int t = 0; t < 4; ++t) {
    int col = c0 + 16 * t + nlo;
    bv[t] = bias[col];
    if (DO_LN) {
      int head = col >> 5, cih = col & 31;
      lwv[t] = lnw[head * DHEAD + cih];
      lbv[t] = lnb[head * DHEAD + cih];
    }
  }

#pragma unroll
  for (int j = 0; j < 8; ++j) {
    int srow = s0 + 16 * w + j + 8 * half;
    if (!DO_LN) {
#pragma unroll
      for (int t = 0; t < 4; ++t) {
        int col = c0 + 16 * t + nlo;
        int head = col >> 5, cih = col & 31;
        float y = acc[t][j] + bv[t];
        out[((size_t)n * SLEN + srow) * CHFULL + head * CCAT + 2 + cih] =
            (_Float16)y;
      }
    } else {
#pragma unroll
      for (int hh = 0; hh < 2; ++hh) {
        float xa = acc[2 * hh + 0][j] + bv[2 * hh + 0];
        float xb = acc[2 * hh + 1][j] + bv[2 * hh + 1];
        float sm = xa + xb;
        float sq = xa * xa + xb * xb;
#pragma unroll
        for (int m = 1; m < 16; m <<= 1) {
          sm += __shfl_xor(sm, m, 16);
          sq += __shfl_xor(sq, m, 16);
        }
        float mu   = sm * (1.0f / 32.0f);
        float var  = sq * (1.0f / 32.0f) - mu * mu;
        float rstd = rsqrtf(var + NORM_EPS);
        float ya = (xa - mu) * rstd * lwv[2 * hh + 0] + lbv[2 * hh + 0];
        float yb = (xb - mu) * rstd * lwv[2 * hh + 1] + lbv[2 * hh + 1];
        int head = ((c0 + 32 * hh) >> 5);
        size_t base = (((size_t)n * NHEAD + head) * SLEN + srow) * CPAD;
        out[base + 2 + nlo]      = (_Float16)ya;
        out[base + 2 + 16 + nlo] = (_Float16)yb;
      }
    }
  }
}

// ---------------------------------------------------------------------------
// ktv: per (n,h), accumulate kcat^T @ vcat over an S-chunk into ktv[48][64].
// block = 1 wave; 3x3 WMMA tiles over padded 48x48, K-dim = s (32/step).
// Padded rows/cols are exactly zero, so atomics are unconditional.
// ---------------------------------------------------------------------------
__global__ __launch_bounds__(32) void ktv_kernel(
    const _Float16* __restrict__ kcat, const _Float16* __restrict__ vcat,
    float* __restrict__ ktv) {
  __shared__ __align__(16) _Float16 kT[CPAD][32];    // [c][s_local]
  __shared__ __align__(16) _Float16 vT[CPAD][32];

  const int lane  = threadIdx.x;
  const int nh    = blockIdx.x;                      // 0..63
  const int chunk = blockIdx.y;                      // 0..15
  const int nlo = lane & 15, half = lane >> 4;
  const int hoff8 = half * 8, hoff16 = half * 16;

  const _Float16* kb = kcat + (size_t)nh * SLEN * CPAD;
  const _Float16* vb = vcat + (size_t)nh * SLEN * CPAD;

  v8f acc[3][3] = {};

  for (int ss = chunk * 512; ss < chunk * 512 + 512; ss += 32) {
    const _Float16* kr = kb + (size_t)(ss + lane) * CPAD;
    const _Float16* vr = vb + (size_t)(ss + lane) * CPAD;
#pragma unroll
    for (int c8 = 0; c8 < 6; ++c8) {
      v8h tk = *(const v8h*)&kr[c8 * 8];
      v8h tv = *(const v8h*)&vr[c8 * 8];
#pragma unroll
      for (int e = 0; e < 8; ++e) {
        kT[c8 * 8 + e][lane] = tk[e];
        vT[c8 * 8 + e][lane] = tv[e];
      }
    }
    __syncthreads();
#pragma unroll
    for (int mt = 0; mt < 3; ++mt) {
      v8h alo = *(const v8h*)&kT[16 * mt + nlo][hoff8];
      v8h ahi = *(const v8h*)&kT[16 * mt + nlo][16 + hoff8];
      v16h a;
#pragma unroll
      for (int i = 0; i < 8; ++i) { a[i] = alo[i]; a[8 + i] = ahi[i]; }
#pragma unroll
      for (int nt = 0; nt < 3; ++nt) {
        v8h blo = *(const v8h*)&vT[16 * nt + nlo][hoff8];
        v8h bhi = *(const v8h*)&vT[16 * nt + nlo][16 + hoff8];
        v16h bb;
#pragma unroll
        for (int i = 0; i < 8; ++i) { bb[i] = blo[i]; bb[8 + i] = bhi[i]; }
        acc[mt][nt] = wmma16(a, bb, acc[mt][nt]);
      }
    }
    __syncthreads();
  }

  float* dst = ktv + (size_t)nh * CPAD * KTVLD;
#pragma unroll
  for (int mt = 0; mt < 3; ++mt)
#pragma unroll
    for (int nt = 0; nt < 3; ++nt)
#pragma unroll
      for (int j = 0; j < 8; ++j) {
        int r  = 16 * mt + j + 8 * half;
        int cc = 16 * nt + nlo;
        unsafeAtomicAdd(&dst[r * KTVLD + cc], acc[mt][nt][j]);
      }
}

// ---------------------------------------------------------------------------
// build Mt[n] (512 x 544, f16), TRANSPOSED: Mt[col][h*34+c] =
//   sum_e (ktv_h[c,e]/S) * fc_W[col, h*34+e]
// block = (n,h); 8 waves, wave w -> cols [64w, 64w+64); K = 64 (padded).
// Transposed layout makes out_gemm's B staging a linear async copy.
// ---------------------------------------------------------------------------
__global__ __launch_bounds__(256) void build_m_kernel(
    const float* __restrict__ ktv, const float* __restrict__ fcW,
    _Float16* __restrict__ Mt) {
  const int n = blockIdx.x, h = blockIdx.y;
  const int tid = threadIdx.x, lane = tid & 31, w = tid >> 5;
  const int nlo = lane & 15, half = lane >> 4;
  const int hoff8 = half * 8, hoff16 = half * 16;
  const float invS = 1.0f / (float)SLEN;
  const float* A = ktv + ((size_t)n * NHEAD + h) * CPAD * KTVLD;
  const int c0 = w * 64;

  for (int mt = 0; mt < 3; ++mt) {
    v16h a[2];
#pragma unroll
    for (int ks = 0; ks < 2; ++ks) {
      int c = 16 * mt + nlo;
#pragma unroll
      for (int jj = 0; jj < 8; ++jj) {
        int kb = ((jj & 3) * 2) + ((jj >> 2) * 16) + hoff8 + ks * 32;
        a[ks][2 * jj + 0] = (_Float16)(A[c * KTVLD + kb + 0] * invS);
        a[ks][2 * jj + 1] = (_Float16)(A[c * KTVLD + kb + 1] * invS);
      }
    }
#pragma unroll
    for (int nt = 0; nt < 4; ++nt) {
      int col = c0 + 16 * nt + nlo;
      v8f acc = {};
#pragma unroll
      for (int ks = 0; ks < 2; ++ks) {
        v16h bb;
#pragma unroll
        for (int i = 0; i < 16; ++i) {
          int e = ks * 32 + hoff8 + i + (i >> 3) * 8;
          int idx = h * CCAT + e;            // B[e][col] = fcW[col][h*34+e]
          if (idx > CHFULL - 1) idx = CHFULL - 1;  // A is 0 there anyway
          bb[i] = (_Float16)fcW[(size_t)col * CHFULL + idx];
        }
        acc = wmma16(a[ks], bb, acc);
      }
#pragma unroll
      for (int j = 0; j < 8; ++j) {
        int c = 16 * mt + j + 8 * half;
        if (c < CCAT)
          Mt[((size_t)n * DHID + col) * CHFULL + h * CCAT + c] =
              (_Float16)acc[j];
      }
    }
  }
}

// ---------------------------------------------------------------------------
// out = qcat[n] (S x 544, f16) @ Mt[n]^T (544 x 512, f16) + fc_b -> fp32
// f16 operands: staged with GLOBAL_LOAD_ASYNC_TO_LDS_B128 (no VGPR pass).
// ---------------------------------------------------------------------------
__global__ __launch_bounds__(256) void out_gemm_kernel(
    const _Float16* __restrict__ qcat, const _Float16* __restrict__ Mt,
    const float* __restrict__ fcb, float* __restrict__ out) {
  __shared__ __align__(16) _Float16 ldsA[128][32];   // [row][k]
  __shared__ __align__(16) _Float16 ldsBt[64][32];   // [col][k]

  const int tid  = threadIdx.x;
  const int lane = tid & 31;
  const int w    = tid >> 5;
  const int nlo  = lane & 15;
  const int half = lane >> 4;
  const int n    = blockIdx.z;
  const int s0   = blockIdx.x * 128;
  const int c0   = blockIdx.y * 64;

  // per-thread async copy assignments
  const int rowA = tid >> 1, ksA = (tid & 1) * 16;   // A: 2 x 16B per thread
  const int colB = tid >> 2, ksB = (tid & 3) * 8;    // B: 1 x 16B per thread
  const _Float16* qrow =
      &qcat[((size_t)n * SLEN + s0 + rowA) * CHFULL + ksA];
  const _Float16* mrow =
      &Mt[((size_t)n * DHID + c0 + colB) * CHFULL + ksB];
  const unsigned ldsAoff =
      (unsigned)(uintptr_t)&ldsA[rowA][ksA];
  const unsigned ldsBoff =
      (unsigned)(uintptr_t)&ldsBt[colB][ksB];

  v8f acc[4] = {};

  for (int k0 = 0; k0 < CHFULL; k0 += 32) {
    async_b128(ldsAoff,      qrow + k0);
    async_b128(ldsAoff + 16, qrow + k0 + 8);
    async_b128(ldsBoff,      mrow + k0);
    wait_async0();
    __syncthreads();

    const int hoff8 = half * 8;
    v8h alo = *(const v8h*)&ldsA[16 * w + nlo][hoff8];
    v8h ahi = *(const v8h*)&ldsA[16 * w + nlo][16 + hoff8];
    v16h a;
#pragma unroll
    for (int i = 0; i < 8; ++i) { a[i] = alo[i]; a[8 + i] = ahi[i]; }
#pragma unroll
    for (int t = 0; t < 4; ++t) {
      v8h blo = *(const v8h*)&ldsBt[16 * t + nlo][half * 8];
      v8h bhi = *(const v8h*)&ldsBt[16 * t + nlo][16 + half * 8];
      v16h bb;
#pragma unroll
      for (int i = 0; i < 8; ++i) { bb[i] = blo[i]; bb[8 + i] = bhi[i]; }
      acc[t] = wmma16(a, bb, acc[t]);
    }
    __syncthreads();
  }

#pragma unroll
  for (int j = 0; j < 8; ++j) {
    int srow = s0 + 16 * w + j + 8 * half;
#pragma unroll
    for (int t = 0; t < 4; ++t) {
      int col = c0 + 16 * t + nlo;
      out[((size_t)n * SLEN + srow) * DHID + col] = acc[t][j] + fcb[col];
    }
  }
}

// ---------------------------------------------------------------------------
extern "C" void kernel_launch(void* const* d_in, const int* in_sizes, int n_in,
                              void* d_out, int out_size, void* d_ws,
                              size_t ws_size, hipStream_t stream) {
  (void)in_sizes; (void)n_in; (void)out_size; (void)ws_size;

  const float* Q    = (const float*)d_in[0];
  const float* K    = (const float*)d_in[1];
  const float* V    = (const float*)d_in[2];
  const float* pos  = (const float*)d_in[3];
  const float* W_Q  = (const float*)d_in[4];
  const float* W_K  = (const float*)d_in[5];
  const float* W_V  = (const float*)d_in[6];
  const float* b_Q  = (const float*)d_in[7];
  const float* b_K  = (const float*)d_in[8];
  const float* b_V  = (const float*)d_in[9];
  const float* lnwK = (const float*)d_in[10];
  const float* lnbK = (const float*)d_in[11];
  const float* lnwV = (const float*)d_in[12];
  const float* lnbV = (const float*)d_in[13];
  const float* fcW  = (const float*)d_in[14];
  const float* fcb  = (const float*)d_in[15];
  float* out = (float*)d_out;

  const size_t QCAT_B = (size_t)NBATCH * SLEN * CHFULL * 2;          // 35.6 MB
  const size_t KCAT_B = (size_t)NBATCH * NHEAD * SLEN * CPAD * 2;    // 50.3 MB
  const size_t KTV_B  = (size_t)NBATCH * NHEAD * CPAD * KTVLD * 4;   // 0.79 MB

  char* ws = (char*)d_ws;
  _Float16* qcat = (_Float16*)ws;
  _Float16* kcat = (_Float16*)(ws + QCAT_B);
  _Float16* vcat = (_Float16*)(ws + QCAT_B + KCAT_B);
  float*    ktv  = (float*)(ws + QCAT_B + 2 * KCAT_B);
  _Float16* Mbuf = (_Float16*)(ws + QCAT_B + 2 * KCAT_B + KTV_B);

  zero_ktv_kernel<<<(NBATCH * NHEAD * CPAD * KTVLD + 255) / 256, 256, 0,
                    stream>>>(ktv);
  init_pos_kernel<<<(NBATCH * SLEN + 255) / 256, 256, 0, stream>>>(
      pos, qcat, kcat, vcat);

  dim3 g(SLEN / 128, DHID / 64, NBATCH);
  proj_kernel<0><<<g, 256, 0, stream>>>(Q, W_Q, b_Q, nullptr, nullptr, qcat);
  proj_kernel<1><<<g, 256, 0, stream>>>(K, W_K, b_K, lnwK, lnbK, kcat);
  proj_kernel<1><<<g, 256, 0, stream>>>(V, W_V, b_V, lnwV, lnbV, vcat);

  ktv_kernel<<<dim3(NBATCH * NHEAD, 16), 32, 0, stream>>>(kcat, vcat, ktv);
  build_m_kernel<<<dim3(NBATCH, NHEAD), 256, 0, stream>>>(ktv, fcW, Mbuf);
  out_gemm_kernel<<<g, 256, 0, stream>>>(qcat, Mbuf, fcb, out);
}
